// Transformer_43843026158057
// MI455X (gfx1250) — compile-verified, matrix operand order checked by reading, not run on hardware
//
#include <hip/hip_runtime.h>
#include <hip/hip_bf16.h>
#include <math.h>
#include <stdint.h>

// ---------------- model constants ----------------
constexpr int D    = 256;
constexpr int NH   = 8;
constexpr int HD   = 32;
constexpr int DFF  = 1024;
constexpr int NL   = 6;
constexpr int NCLS = 91;
constexpr int NPAT = 3;
constexpr int NPOS = 300;
constexpr int BS   = 2;
constexpr int H    = 40;
constexpr int W    = 40;
constexpr int LE   = H * W;       // 1600 encoder tokens per batch
constexpr int LQ   = NPAT * NPOS; // 900 decoder queries per batch
constexpr int BL   = BS * LE;     // 3200
constexpr int BQ   = BS * LQ;     // 1800
constexpr float ATT_SCALE = 0.17677669529663687f; // 32^-0.5
constexpr float TWO_PI = 6.283185307179586f;

// ---------------- CDNA5 vector types ----------------
typedef _Float16 v16h __attribute__((ext_vector_type(16)));
typedef _Float16 v8h  __attribute__((ext_vector_type(8)));
typedef _Float16 v4h  __attribute__((ext_vector_type(4)));
typedef float    v8f  __attribute__((ext_vector_type(8)));
typedef unsigned int u32x4 __attribute__((ext_vector_type(4)));
typedef int          i32x4 __attribute__((ext_vector_type(4)));
typedef int          i32x8 __attribute__((ext_vector_type(8)));

// ============================================================
// GEMM: C[M,N] = act(A[M,K] @ Wh[N,K]^T + bias[N])
// A fp32 (vectorized b128 staging + cvt), Wh f16 (TDM tensor_load_to_lds),
// f16 WMMA with f32 accumulate. Block tile 128x128, 8 waves (wave32),
// wave tile 32x64 = 2x4 v_wmma_f32_16x16x32_f16 fragments.
// ============================================================
#define TM 128
#define TN 128
#define TK 32

__global__ __launch_bounds__(256)
void gemm_wmma_kernel(const float* __restrict__ A, const _Float16* __restrict__ Wh,
                      const float* __restrict__ bias, float* __restrict__ C,
                      int M, int N, int K, int act)
{
    __shared__ __align__(16) _Float16 As[TM][TK];
    __shared__ __align__(16) _Float16 Bs[TN][TK];

    const int tid  = threadIdx.x;
    const int lane = tid & 31;
    const int wid  = tid >> 5;   // 0..7
    const int wm   = wid & 3;    // 4 wave rows (32 rows each)
    const int wn   = wid >> 2;   // 2 wave cols (64 cols each)
    const int m0   = blockIdx.x * TM;
    const int n0   = blockIdx.y * TN;

    v8f acc[2][4] = {};

    const int lrow = lane & 15;
    const int lk8  = (lane >> 4) * 8;  // K sub-offset per ISA fragment layout

    for (int k0 = 0; k0 < K; k0 += TK) {
        // ---- B tile via Tensor Data Mover: 128 rows x 32 cols of f16 ----
        if (tid == 0) {
            unsigned lds_addr = (unsigned)(uintptr_t)(&Bs[0][0]);
            unsigned long long gaddr =
                (unsigned long long)(uintptr_t)(Wh + (size_t)n0 * K + k0);
            unsigned t0 = (unsigned)(K - k0);   // remaining tensor extent dim0 (K)
            unsigned t1 = (unsigned)(N - n0);   // remaining tensor extent dim1 (N rows)
            u32x4 g0 = { 1u,                                   // count=1, user mode
                         lds_addr,                             // lds_addr [63:32]
                         (unsigned)(gaddr & 0xffffffffu),      // global_addr [95:64]
                         (unsigned)((gaddr >> 32) & 0x01ffffffu) | (2u << 30) }; // +type=2
            i32x8 g1 = { (int)(1u << 16),                      // data_size=1 (2 bytes)
                         (int)((t0 & 0xffffu) << 16),          // tensor_dim0 lo16
                         (int)(((t0 >> 16) & 0xffffu) | ((t1 & 0xffffu) << 16)),
                         (int)(((t1 >> 16) & 0xffffu) | (32u << 16)),  // tile_dim0=32
                         (int)128u,                            // tile_dim1=128
                         (int)(unsigned)K,                     // tensor_dim0_stride lo32
                         0, 0 };
            i32x4 g2 = { 0, 0, 0, 0 };
            i32x4 g3 = { 0, 0, 0, 0 };
            i32x8 g4 = { 0, 0, 0, 0, 0, 0, 0, 0 };
            __builtin_amdgcn_tensor_load_to_lds(g0, g1, g2, g3, g4, 0);
            __builtin_amdgcn_s_wait_tensorcnt(0);
        }
        // prefetch next A K-tile (global_prefetch_b8)
        if (k0 + TK < K) {
            int r = tid >> 1, c = (tid & 1) * 16;
            if (m0 + r < M) __builtin_prefetch(&A[(size_t)(m0 + r) * K + k0 + TK + c], 0, 1);
        }
        // ---- A tile: 128x32 fp32, 4 floats/thread/pass, cvt -> f16 ----
        #pragma unroll
        for (int i = 0; i < 4; ++i) {
            int idx = tid + i * 256;           // 0..1023 float4 slots
            int r = idx >> 3, c = (idx & 7) * 4;
            int gm = m0 + r;
            float4 v = make_float4(0.f, 0.f, 0.f, 0.f);
            if (gm < M) v = *(const float4*)&A[(size_t)gm * K + k0 + c];
            v4h h = { (_Float16)v.x, (_Float16)v.y, (_Float16)v.z, (_Float16)v.w };
            *(v4h*)&As[r][c] = h;
        }
        __syncthreads();

        v16h afr[2], bfr[4];
        #pragma unroll
        for (int i = 0; i < 2; ++i) {
            const _Float16* p = &As[wm * 32 + i * 16 + lrow][lk8];
            v8h lo = *(const v8h*)p;
            v8h hi = *(const v8h*)(p + 16);
            afr[i] = __builtin_shufflevector(lo, hi, 0,1,2,3,4,5,6,7,8,9,10,11,12,13,14,15);
        }
        #pragma unroll
        for (int j = 0; j < 4; ++j) {
            const _Float16* p = &Bs[wn * 64 + j * 16 + lrow][lk8];
            v8h lo = *(const v8h*)p;
            v8h hi = *(const v8h*)(p + 16);
            bfr[j] = __builtin_shufflevector(lo, hi, 0,1,2,3,4,5,6,7,8,9,10,11,12,13,14,15);
        }
        #pragma unroll
        for (int i = 0; i < 2; ++i)
            #pragma unroll
            for (int j = 0; j < 4; ++j)
                acc[i][j] = __builtin_amdgcn_wmma_f32_16x16x32_f16(
                    false, afr[i], false, bfr[j], (short)0, acc[i][j], false, false);
        __syncthreads();
    }

    // store per C/D layout: lanes 0-15 -> M 0..7, lanes 16-31 -> M 8..15
    const int mo = (lane >> 4) * 8;
    const int nn = lane & 15;
    #pragma unroll
    for (int i = 0; i < 2; ++i)
        #pragma unroll
        for (int j = 0; j < 4; ++j) {
            int nb = n0 + wn * 64 + j * 16 + nn;
            if (nb >= N) continue;
            float bv = bias ? bias[nb] : 0.f;
            int mb = m0 + wm * 32 + i * 16 + mo;
            #pragma unroll
            for (int r = 0; r < 8; ++r) {
                int m = mb + r;
                if (m < M) {
                    float v = acc[i][j][r] + bv;
                    if (act) v = fmaxf(v, 0.f);
                    C[(size_t)m * N + nb] = v;
                }
            }
        }
}

// f32 -> f16 weight conversion (vectorized x4; n divisible by 4)
__global__ __launch_bounds__(256)
void f32_to_f16_kernel(_Float16* __restrict__ out, const float* __restrict__ in, int n4)
{
    int i = blockIdx.x * 256 + threadIdx.x;
    if (i >= n4) return;
    float4 v = *(const float4*)&in[(size_t)i * 4];
    v4h h = { (_Float16)v.x, (_Float16)v.y, (_Float16)v.z, (_Float16)v.w };
    *(v4h*)&out[(size_t)i * 4] = h;
}

// ============================================================
// LayerNorm(x + res) * gamma + beta ; one block (256 thr) per row
// ============================================================
__global__ __launch_bounds__(256)
void ln_kernel(float* __restrict__ y, const float* __restrict__ x,
               const float* __restrict__ res, const float* __restrict__ gb, int rows)
{
    __shared__ float red[256];
    __shared__ float s_mean, s_rstd;
    int row = blockIdx.x;
    int t = threadIdx.x;
    float v = x[(size_t)row * D + t] + (res ? res[(size_t)row * D + t] : 0.f);
    red[t] = v; __syncthreads();
    for (int s = 128; s > 0; s >>= 1) { if (t < s) red[t] += red[t + s]; __syncthreads(); }
    if (t == 0) s_mean = red[0] * (1.f / D);
    __syncthreads();
    float c = v - s_mean;
    red[t] = c * c; __syncthreads();
    for (int s = 128; s > 0; s >>= 1) { if (t < s) red[t] += red[t + s]; __syncthreads(); }
    if (t == 0) s_rstd = rsqrtf(red[0] * (1.f / D) + 1e-5f);
    __syncthreads();
    y[(size_t)row * D + t] = c * s_rstd * gb[t] + gb[D + t];
}

// ============================================================
// pos features: cumsum-normalized row/col positions -> sin/cos p1d
// ============================================================
__global__ __launch_bounds__(256)
void posfeat_rc_kernel(float* __restrict__ feat_row, float* __restrict__ feat_col,
                       const unsigned char* __restrict__ mask)
{
    __shared__ float prow[W], pcol[H];
    int b = blockIdx.x, t = threadIdx.x;
    if (t == 0) {
        float acc = 0.f;
        for (int x = 0; x < W; ++x) { acc += mask[b * H * W + x] ? 0.f : 1.f; prow[x] = acc; }
        float inv = 1.f / acc;
        for (int x = 0; x < W; ++x) prow[x] = (prow[x] - 0.5f) * inv;
        acc = 0.f;
        for (int y = 0; y < H; ++y) { acc += mask[b * H * W + y * W] ? 0.f : 1.f; pcol[y] = acc; }
        inv = 1.f / acc;
        for (int y = 0; y < H; ++y) pcol[y] = (pcol[y] - 0.5f) * inv;
    }
    __syncthreads();
    int j = t >> 1;
    float dt = powf(10000.f, (float)j * (1.f / 128.f));
    float w = TWO_PI / dt;
    for (int x = 0; x < W; ++x) {
        float v = prow[x] * w;
        feat_row[((size_t)b * W + x) * D + t] = (t & 1) ? cosf(v) : sinf(v);
    }
    for (int y = 0; y < H; ++y) {
        float v = pcol[y] * w;
        feat_col[((size_t)b * H + y) * D + t] = (t & 1) ? cosf(v) : sinf(v);
    }
}

__global__ __launch_bounds__(256)
void p1d_kernel(float* __restrict__ out, const float* __restrict__ pos)
{
    int r = blockIdx.x, t = threadIdx.x;
    int j = t >> 1;
    float dt = powf(10000.f, (float)j * (1.f / 128.f));
    float v = pos[r] * TWO_PI / dt;
    out[(size_t)r * D + t] = (t & 1) ? cosf(v) : sinf(v);
}

__global__ __launch_bounds__(256)
void p2d_kernel(float* __restrict__ out, const float* __restrict__ rx,
                const float* __restrict__ ry)
{
    int r = blockIdx.x, t = threadIdx.x;
    float p; int i;
    if (t < 128) { p = ry[r]; i = t; } else { p = rx[r]; i = t - 128; }
    int j = i >> 1;
    float dt = powf(10000.f, (float)j * (1.f / 64.f));
    float v = p * TWO_PI / dt;
    out[(size_t)r * D + t] = (i & 1) ? cosf(v) : sinf(v);
}

__global__ void init_ref_kernel(float* __restrict__ refx, float* __restrict__ refy,
                                float* __restrict__ refinv, const float* __restrict__ position)
{
    int i = blockIdx.x * 256 + threadIdx.x;
    if (i >= BQ) return;
    int l = i % LQ, p = l % NPOS;
    float x = position[p * 2 + 0], y = position[p * 2 + 1];
    refx[i] = x; refy[i] = y;
    float cx = fminf(fmaxf(x, 0.f), 1.f);
    float cy = fminf(fmaxf(y, 0.f), 1.f);
    refinv[i * 2 + 0] = logf(fmaxf(cx, 1e-5f) / fmaxf(1.f - cx, 1e-5f));
    refinv[i * 2 + 1] = logf(fmaxf(cy, 1e-5f) / fmaxf(1.f - cy, 1e-5f));
}

__global__ __launch_bounds__(256)
void init_tgt_kernel(float* __restrict__ tgt, const float* __restrict__ pattern)
{
    int row = blockIdx.x, t = threadIdx.x;
    int l = row % LQ;
    tgt[(size_t)row * D + t] = pattern[(l / NPOS) * D + t];
}

__global__ __launch_bounds__(256)
void nchw_to_nhwc_kernel(float* __restrict__ out, const float* __restrict__ in)
{
    size_t i = (size_t)blockIdx.x * 256 + threadIdx.x;
    int c = i & (D - 1);
    size_t r = i >> 8;
    int x = r % W, y = (r / W) % H, b = r / (H * W);
    out[i] = in[(((size_t)b * D + c) * H + y) * W + x];
}

__global__ __launch_bounds__(256)
void add_pe_kernel(float* __restrict__ out, const float* __restrict__ s,
                   const float* __restrict__ pe, int axis)
{
    size_t i = (size_t)blockIdx.x * 256 + threadIdx.x;
    int c = i & (D - 1);
    size_t r = i >> 8;
    int x = r % W, y = (r / W) % H, b = r / (H * W);
    int k = axis ? y : x;
    out[i] = s[i] + pe[((size_t)b * 40 + k) * D + c];
}

__global__ void add_kernel(float* __restrict__ out, const float* __restrict__ a,
                           const float* __restrict__ b, int n)
{
    int i = blockIdx.x * 256 + threadIdx.x;
    if (i < n) out[i] = a[i] + b[i];
}

__global__ __launch_bounds__(256)
void pool_h_kernel(float* __restrict__ out, const float* __restrict__ in)
{
    int c = threadIdx.x, x = blockIdx.x % W, b = blockIdx.x / W;
    float s = 0.f;
    for (int y = 0; y < H; ++y) s += in[(((size_t)b * H + y) * W + x) * D + c];
    out[((size_t)b * W + x) * D + c] = s * (1.f / H);
}

__global__ __launch_bounds__(256)
void pool_w_kernel(float* __restrict__ out, const float* __restrict__ in)
{
    int c = threadIdx.x, y = blockIdx.x % H, b = blockIdx.x / H;
    float s = 0.f;
    for (int x = 0; x < W; ++x) s += in[(((size_t)b * H + y) * W + x) * D + c];
    out[((size_t)b * H + y) * D + c] = s * (1.f / W);
}

// RCDA scores + mask + softmax over 40 keys ; grid (L, B*NH), block 64
__global__ __launch_bounds__(64)
void rcda_scores_kernel(float* __restrict__ att, const float* __restrict__ q,
                        const float* __restrict__ kp, const unsigned char* __restrict__ mask,
                        int L, int mstride)
{
    __shared__ float sc[40];
    __shared__ float sinv;
    int l = blockIdx.x, bn = blockIdx.y;
    int n = bn % NH, b = bn / NH;
    int t = threadIdx.x;
    if (t < 40) {
        const float* qp = q  + ((size_t)b * L + l) * D + n * HD;
        const float* kk = kp + ((size_t)b * 40 + t) * D + n * HD;
        float s = 0.f;
        for (int d = 0; d < HD; ++d) s += qp[d] * kk[d];
        s *= ATT_SCALE;
        if (mask[b * H * W + t * mstride]) s = -1e30f;
        sc[t] = s;
    }
    __syncthreads();
    if (t == 0) {
        float m = -1e30f;
        for (int i = 0; i < 40; ++i) m = fmaxf(m, sc[i]);
        float su = 0.f;
        for (int i = 0; i < 40; ++i) { float e = __expf(sc[i] - m); sc[i] = e; su += e; }
        sinv = 1.f / su;
    }
    __syncthreads();
    if (t < 40) att[(((size_t)b * NH + n) * L + l) * 40 + t] = sc[t] * sinv;
}

// RCDA apply: o[b,l,n,d] = sum_h ac[h] * sum_w ar[w] * vv[b,h,w,n,d]
__global__ __launch_bounds__(64)
void rcda_apply_kernel(float* __restrict__ out, const float* __restrict__ ar,
                       const float* __restrict__ ac, const float* __restrict__ vv, int L)
{
    __shared__ float sar[40], sac[40];
    int l = blockIdx.x, bn = blockIdx.y;
    int n = bn % NH, b = bn / NH;
    int t = threadIdx.x;
    if (t < 40) {
        sar[t] = ar[(((size_t)b * NH + n) * L + l) * 40 + t];
        sac[t] = ac[(((size_t)b * NH + n) * L + l) * 40 + t];
    }
    __syncthreads();
    if (t < HD) {
        float o = 0.f;
        for (int y = 0; y < H; ++y) {
            const float* vy = vv + ((((size_t)b * H + y) * W) * NH + n) * HD + t;
            float inner = 0.f;
            for (int x = 0; x < W; ++x) inner += sar[x] * vy[(size_t)x * NH * HD];
            o += sac[y] * inner;
        }
        out[((size_t)b * L + l) * D + n * HD + t] = o;
    }
}

// Decoder self-attention, fused softmax ; grid (LQ, B*NH), block 128
__global__ __launch_bounds__(128)
void mha_kernel(float* __restrict__ out, const float* __restrict__ qk,
                const float* __restrict__ v)
{
    __shared__ float p[LQ];
    __shared__ float red[128];
    __shared__ float osum[4][HD];
    int qrow = blockIdx.x, bn = blockIdx.y;
    int n = bn % NH, b = bn / NH;
    int t = threadIdx.x;
    const float* qp = qk + ((size_t)(b * LQ + qrow)) * 512 + n * HD;
    float lmax = -1e30f;
    for (int k = t; k < LQ; k += 128) {
        const float* kp = qk + ((size_t)(b * LQ + k)) * 512 + 256 + n * HD;
        float s = 0.f;
        for (int d = 0; d < HD; ++d) s += qp[d] * kp[d];
        s *= ATT_SCALE;
        p[k] = s;
        lmax = fmaxf(lmax, s);
    }
    red[t] = lmax; __syncthreads();
    for (int s = 64; s > 0; s >>= 1) { if (t < s) red[t] = fmaxf(red[t], red[t + s]); __syncthreads(); }
    float m = red[0]; __syncthreads();
    float lsum = 0.f;
    for (int k = t; k < LQ; k += 128) { float e = __expf(p[k] - m); p[k] = e; lsum += e; }
    red[t] = lsum; __syncthreads();
    for (int s = 64; s > 0; s >>= 1) { if (t < s) red[t] += red[t + s]; __syncthreads(); }
    float inv = 1.f / red[0]; __syncthreads();
    int d = t & 31, part = t >> 5;
    float o = 0.f;
    for (int k = part; k < LQ; k += 4) o += p[k] * v[((size_t)(b * LQ + k)) * D + n * HD + d];
    osum[part][d] = o; __syncthreads();
    if (t < HD)
        out[((size_t)b * LQ + qrow) * D + n * HD + t] =
            (osum[0][t] + osum[1][t] + osum[2][t] + osum[3][t]) * inv;
}

__global__ void bbox_out_kernel(float* __restrict__ out, const float* __restrict__ bb,
                                const float* __restrict__ refinv)
{
    int i = blockIdx.x * 256 + threadIdx.x;
    if (i >= BQ * 4) return;
    int c = i & 3, r = i >> 2;
    float v = bb[i];
    if (c < 2) v += refinv[r * 2 + c];
    out[i] = 1.f / (1.f + __expf(-v));
}

// ---------------- host-side helpers ----------------
static void gemm(const float* A, const _Float16* Wh, const float* bias, float* C,
                 int M, int N, int K, int act, hipStream_t s)
{
    dim3 g((M + TM - 1) / TM, (N + TN - 1) / TN);
    gemm_wmma_kernel<<<g, 256, 0, s>>>(A, Wh, bias, C, M, N, K, act);
}
static void ln(float* y, const float* x, const float* res, const float* gb, int rows, hipStream_t s)
{
    ln_kernel<<<rows, 256, 0, s>>>(y, x, res, gb, rows);
}
static void conv16(_Float16* out, const float* in, size_t n, hipStream_t s)
{
    int n4 = (int)(n / 4);
    f32_to_f16_kernel<<<(n4 + 255) / 256, 256, 0, s>>>(out, in, n4);
}

extern "C" void kernel_launch(void* const* d_in, const int* in_sizes, int n_in,
                              void* d_out, int out_size, void* d_ws, size_t ws_size,
                              hipStream_t stream)
{
    // -------- inputs (setup_inputs order) --------
    const float* srcs       = (const float*)d_in[0];
    const float* enc_qkv_w  = (const float*)d_in[1];
    const float* enc_qkv_b  = (const float*)d_in[2];
    const float* enc_out_w  = (const float*)d_in[3];
    const float* enc_out_b  = (const float*)d_in[4];
    const float* enc_ln1    = (const float*)d_in[5];
    const float* enc_ffn_w1 = (const float*)d_in[6];
    const float* enc_ffn_b1 = (const float*)d_in[7];
    const float* enc_ffn_w2 = (const float*)d_in[8];
    const float* enc_ffn_b2 = (const float*)d_in[9];
    const float* enc_ln2    = (const float*)d_in[10];
    const float* dec_sa_w   = (const float*)d_in[11];
    const float* dec_sa_b   = (const float*)d_in[12];
    const float* dec_sa_ow  = (const float*)d_in[13];
    const float* dec_sa_ob  = (const float*)d_in[14];
    const float* dec_ln2    = (const float*)d_in[15];
    const float* dec_ca_w   = (const float*)d_in[16];
    const float* dec_ca_b   = (const float*)d_in[17];
    const float* dec_ca_ow  = (const float*)d_in[18];
    const float* dec_ca_ob  = (const float*)d_in[19];
    const float* dec_ln1    = (const float*)d_in[20];
    const float* dec_ffn_w1 = (const float*)d_in[21];
    const float* dec_ffn_b1 = (const float*)d_in[22];
    const float* dec_ffn_w2 = (const float*)d_in[23];
    const float* dec_ffn_b2 = (const float*)d_in[24];
    const float* dec_ln3    = (const float*)d_in[25];
    const float* pattern    = (const float*)d_in[26];
    const float* position   = (const float*)d_in[27];
    const float* a2w1       = (const float*)d_in[28];
    const float* a2b1       = (const float*)d_in[29];
    const float* a2w2       = (const float*)d_in[30];
    const float* a2b2       = (const float*)d_in[31];
    const float* a1w1       = (const float*)d_in[32];
    const float* a1b1       = (const float*)d_in[33];
    const float* a1w2       = (const float*)d_in[34];
    const float* a1b2       = (const float*)d_in[35];
    const float* cls_w      = (const float*)d_in[36];
    const float* cls_b      = (const float*)d_in[37];
    const float* bbox_w1    = (const float*)d_in[38];
    const float* bbox_b1    = (const float*)d_in[39];
    const float* bbox_w2    = (const float*)d_in[40];
    const float* bbox_b2    = (const float*)d_in[41];
    const float* bbox_w3    = (const float*)d_in[42];
    const float* bbox_b3    = (const float*)d_in[43];
    const unsigned char* masks = (const unsigned char*)d_in[44];

    float* outp = (float*)d_out;
    float* cls_out   = outp;                              // (NL,BS,LQ,NCLS)
    float* coord_out = outp + (size_t)NL * BQ * NCLS;     // (NL,BS,LQ,4)

    // -------- fp32 workspace --------
    float* ws = (float*)d_ws;
    size_t off = 0;
    auto alloc = [&](size_t n) { float* p = ws + off; off += (n + 63) & ~(size_t)63; return p; };
    float* S    = alloc((size_t)BL * D);     // activations (B,H,W,D); becomes `mem`
    float* KR   = alloc((size_t)BL * D);
    float* KC   = alloc((size_t)BL * D);
    float* QR   = alloc((size_t)BL * D);
    float* QC   = alloc((size_t)BL * D);
    float* VV   = alloc((size_t)BL * D);
    float* O    = alloc((size_t)BL * D);
    float* S2   = alloc((size_t)BL * D);
    float* TMP1 = alloc((size_t)BL * DFF);   // FFN mid / generic scratch
    float* AR   = alloc((size_t)BS * NH * LE * 40);
    float* AC   = alloc((size_t)BS * NH * LE * 40);
    float* PF_ROW = alloc((size_t)BS * 40 * D);
    float* PF_COL = alloc((size_t)BS * 40 * D);
    float* PE_ROW = alloc((size_t)BS * 40 * D);
    float* PE_COL = alloc((size_t)BS * 40 * D);
    float* PMID   = alloc((size_t)BS * 40 * D);
    float* KRP    = alloc((size_t)BS * 40 * D);
    float* KCP    = alloc((size_t)BS * 40 * D);
    float* TGT    = alloc((size_t)BQ * D);
    float* QBUF   = alloc((size_t)BQ * D);
    float* QKBUF  = alloc((size_t)BQ * 512);
    float* VBUF   = alloc((size_t)BQ * D);
    float* ATT    = alloc((size_t)BQ * D);
    float* QPOS   = alloc((size_t)BQ * D);
    float* QPX    = alloc((size_t)BQ * D);
    float* QPY    = alloc((size_t)BQ * D);
    float* PDF    = alloc((size_t)BQ * D);
    float* REFX   = alloc(BQ);
    float* REFY   = alloc(BQ);
    float* REFINV = alloc((size_t)BQ * 2);
    float* BB1    = alloc((size_t)BQ * D);
    float* BB2    = alloc((size_t)BQ * D);
    float* BB3    = alloc((size_t)BQ * 4);

    // -------- f16 weight workspace (converted once per launch) --------
    _Float16* hws = (_Float16*)(ws + off);
    size_t hoff = 0;
    auto halloc = [&](size_t n) { _Float16* p = hws + hoff; hoff += (n + 63) & ~(size_t)63; return p; };
    _Float16* H_enc_qkv  = halloc((size_t)NL * 5 * D * D);
    _Float16* H_enc_out  = halloc((size_t)NL * D * D);
    _Float16* H_enc_f1   = halloc((size_t)NL * DFF * D);
    _Float16* H_enc_f2   = halloc((size_t)NL * D * DFF);
    _Float16* H_dec_sa   = halloc((size_t)NL * 3 * D * D);
    _Float16* H_dec_saow = halloc((size_t)NL * D * D);
    _Float16* H_dec_ca   = halloc((size_t)NL * 5 * D * D);
    _Float16* H_dec_caow = halloc((size_t)NL * D * D);
    _Float16* H_dec_f1   = halloc((size_t)NL * DFF * D);
    _Float16* H_dec_f2   = halloc((size_t)NL * D * DFF);
    _Float16* H_a2w1     = halloc((size_t)D * D);
    _Float16* H_a2w2     = halloc((size_t)D * D);
    _Float16* H_a1w1     = halloc((size_t)D * D);
    _Float16* H_a1w2     = halloc((size_t)D * D);
    _Float16* H_cls      = halloc((size_t)NCLS * D);
    _Float16* H_bb1      = halloc((size_t)D * D);
    _Float16* H_bb2      = halloc((size_t)D * D);
    _Float16* H_bb3      = halloc((size_t)4 * D);
    (void)ws_size; (void)in_sizes; (void)n_in; (void)out_size;

    conv16(H_enc_qkv, enc_qkv_w, (size_t)NL * 5 * D * D, stream);
    conv16(H_enc_out, enc_out_w, (size_t)NL * D * D, stream);
    conv16(H_enc_f1, enc_ffn_w1, (size_t)NL * DFF * D, stream);
    conv16(H_enc_f2, enc_ffn_w2, (size_t)NL * D * DFF, stream);
    conv16(H_dec_sa, dec_sa_w, (size_t)NL * 3 * D * D, stream);
    conv16(H_dec_saow, dec_sa_ow, (size_t)NL * D * D, stream);
    conv16(H_dec_ca, dec_ca_w, (size_t)NL * 5 * D * D, stream);
    conv16(H_dec_caow, dec_ca_ow, (size_t)NL * D * D, stream);
    conv16(H_dec_f1, dec_ffn_w1, (size_t)NL * DFF * D, stream);
    conv16(H_dec_f2, dec_ffn_w2, (size_t)NL * D * DFF, stream);
    conv16(H_a2w1, a2w1, (size_t)D * D, stream);
    conv16(H_a2w2, a2w2, (size_t)D * D, stream);
    conv16(H_a1w1, a1w1, (size_t)D * D, stream);
    conv16(H_a1w2, a1w2, (size_t)D * D, stream);
    conv16(H_cls, cls_w, (size_t)NCLS * D, stream);
    conv16(H_bb1, bbox_w1, (size_t)D * D, stream);
    conv16(H_bb2, bbox_w2, (size_t)D * D, stream);
    conv16(H_bb3, bbox_w3, (size_t)4 * D, stream);

    const int NELT = BL * D;  // 819200, divisible by 256

    // -------- setup --------
    nchw_to_nhwc_kernel<<<NELT / 256, 256, 0, stream>>>(S, srcs);
    posfeat_rc_kernel<<<BS, 256, 0, stream>>>(PF_ROW, PF_COL, masks);
    gemm(PF_ROW, H_a1w1, a1b1, PMID, BS * 40, D, D, 1, stream);
    gemm(PMID, H_a1w2, a1b2, PE_ROW, BS * 40, D, D, 0, stream);
    gemm(PF_COL, H_a1w1, a1b1, PMID, BS * 40, D, D, 1, stream);
    gemm(PMID, H_a1w2, a1b2, PE_COL, BS * 40, D, D, 0, stream);

    init_ref_kernel<<<(BQ + 255) / 256, 256, 0, stream>>>(REFX, REFY, REFINV, position);
    init_tgt_kernel<<<BQ, 256, 0, stream>>>(TGT, pattern);
    p2d_kernel<<<BQ, 256, 0, stream>>>(PDF, REFX, REFY);
    gemm(PDF, H_a2w1, a2b1, TMP1, BQ, D, D, 1, stream);
    gemm(TMP1, H_a2w2, a2b2, QPOS, BQ, D, D, 0, stream);
    p1d_kernel<<<BQ, 256, 0, stream>>>(PDF, REFX);
    gemm(PDF, H_a1w1, a1b1, TMP1, BQ, D, D, 1, stream);
    gemm(TMP1, H_a1w2, a1b2, QPX, BQ, D, D, 0, stream);
    p1d_kernel<<<BQ, 256, 0, stream>>>(PDF, REFY);
    gemm(PDF, H_a1w1, a1b1, TMP1, BQ, D, D, 1, stream);
    gemm(TMP1, H_a1w2, a1b2, QPY, BQ, D, D, 0, stream);

    // -------- encoder --------
    for (int i = 0; i < NL; ++i) {
        const _Float16* qw = H_enc_qkv + (size_t)i * 5 * D * D;
        const float*    qb = enc_qkv_b + (size_t)i * 5 * D;
        add_pe_kernel<<<NELT / 256, 256, 0, stream>>>(KR, S, PE_ROW, 0);
        add_pe_kernel<<<NELT / 256, 256, 0, stream>>>(KC, S, PE_COL, 1);
        gemm(KR, qw + 0 * D * D, qb + 0 * D, QR, BL, D, D, 0, stream);
        gemm(KC, qw + 1 * D * D, qb + 1 * D, QC, BL, D, D, 0, stream);
        gemm(KR, qw + 2 * D * D, qb + 2 * D, TMP1, BL, D, D, 0, stream);
        pool_h_kernel<<<BS * W, 256, 0, stream>>>(KRP, TMP1);
        gemm(KC, qw + 3 * D * D, qb + 3 * D, TMP1, BL, D, D, 0, stream);
        pool_w_kernel<<<BS * H, 256, 0, stream>>>(KCP, TMP1);
        gemm(S, qw + 4 * D * D, qb + 4 * D, VV, BL, D, D, 0, stream);
        rcda_scores_kernel<<<dim3(LE, BS * NH), 64, 0, stream>>>(AR, QR, KRP, masks, LE, 1);
        rcda_scores_kernel<<<dim3(LE, BS * NH), 64, 0, stream>>>(AC, QC, KCP, masks, LE, W);
        rcda_apply_kernel<<<dim3(LE, BS * NH), 64, 0, stream>>>(O, AR, AC, VV, LE);
        gemm(O, H_enc_out + (size_t)i * D * D, enc_out_b + (size_t)i * D, S2, BL, D, D, 0, stream);
        ln(S, S, S2, enc_ln1 + (size_t)i * 2 * D, BL, stream);
        gemm(S, H_enc_f1 + (size_t)i * DFF * D, enc_ffn_b1 + (size_t)i * DFF, TMP1, BL, DFF, D, 1, stream);
        gemm(TMP1, H_enc_f2 + (size_t)i * D * DFF, enc_ffn_b2 + (size_t)i * D, S2, BL, D, DFF, 0, stream);
        ln(S, S, S2, enc_ln2 + (size_t)i * 2 * D, BL, stream);
    }

    // mem = S ; decoder cross-attn keys
    add_pe_kernel<<<NELT / 256, 256, 0, stream>>>(KR, S, PE_ROW, 0);
    add_pe_kernel<<<NELT / 256, 256, 0, stream>>>(KC, S, PE_COL, 1);

    // -------- decoder --------
    for (int i = 0; i < NL; ++i) {
        // self-attention
        add_kernel<<<(BQ * D + 255) / 256, 256, 0, stream>>>(QBUF, TGT, QPOS, BQ * D);
        gemm(QBUF, H_dec_sa + (size_t)i * 3 * D * D, dec_sa_b + (size_t)i * 3 * D,
             QKBUF, BQ, 2 * D, D, 0, stream);
        gemm(TGT, H_dec_sa + (size_t)i * 3 * D * D + (size_t)2 * D * D,
             dec_sa_b + (size_t)i * 3 * D + 2 * D, VBUF, BQ, D, D, 0, stream);
        mha_kernel<<<dim3(LQ, BS * NH), 128, 0, stream>>>(ATT, QKBUF, VBUF);
        gemm(ATT, H_dec_saow + (size_t)i * D * D, dec_sa_ob + (size_t)i * D, S2, BQ, D, D, 0, stream);
        ln(TGT, TGT, S2, dec_ln2 + (size_t)i * 2 * D, BQ, stream);

        // cross RCDA
        const _Float16* cw = H_dec_ca + (size_t)i * 5 * D * D;
        const float*    cb = dec_ca_b + (size_t)i * 5 * D;
        add_kernel<<<(BQ * D + 255) / 256, 256, 0, stream>>>(QBUF, TGT, QPX, BQ * D);
        gemm(QBUF, cw + 0 * D * D, cb + 0 * D, QR, BQ, D, D, 0, stream);
        add_kernel<<<(BQ * D + 255) / 256, 256, 0, stream>>>(QBUF, TGT, QPY, BQ * D);
        gemm(QBUF, cw + 1 * D * D, cb + 1 * D, QC, BQ, D, D, 0, stream);
        gemm(KR, cw + 2 * D * D, cb + 2 * D, TMP1, BL, D, D, 0, stream);
        pool_h_kernel<<<BS * W, 256, 0, stream>>>(KRP, TMP1);
        gemm(KC, cw + 3 * D * D, cb + 3 * D, TMP1, BL, D, D, 0, stream);
        pool_w_kernel<<<BS * H, 256, 0, stream>>>(KCP, TMP1);
        gemm(S, cw + 4 * D * D, cb + 4 * D, VV, BL, D, D, 0, stream);
        rcda_scores_kernel<<<dim3(LQ, BS * NH), 64, 0, stream>>>(AR, QR, KRP, masks, LQ, 1);
        rcda_scores_kernel<<<dim3(LQ, BS * NH), 64, 0, stream>>>(AC, QC, KCP, masks, LQ, W);
        rcda_apply_kernel<<<dim3(LQ, BS * NH), 64, 0, stream>>>(O, AR, AC, VV, LQ);
        gemm(O, H_dec_caow + (size_t)i * D * D, dec_ca_ob + (size_t)i * D, S2, BQ, D, D, 0, stream);
        ln(TGT, TGT, S2, dec_ln1 + (size_t)i * 2 * D, BQ, stream);

        // FFN
        gemm(TGT, H_dec_f1 + (size_t)i * DFF * D, dec_ffn_b1 + (size_t)i * DFF, TMP1, BQ, DFF, D, 1, stream);
        gemm(TMP1, H_dec_f2 + (size_t)i * D * DFF, dec_ffn_b2 + (size_t)i * D, S2, BQ, D, DFF, 0, stream);
        ln(TGT, TGT, S2, dec_ln3 + (size_t)i * 2 * D, BQ, stream);

        // heads
        gemm(TGT, H_cls, cls_b, cls_out + (size_t)i * BQ * NCLS, BQ, NCLS, D, 0, stream);
        gemm(TGT, H_bb1, bbox_b1, BB1, BQ, D, D, 1, stream);
        gemm(BB1, H_bb2, bbox_b2, BB2, BQ, D, D, 1, stream);
        gemm(BB2, H_bb3, bbox_b3, BB3, BQ, 4, D, 0, stream);
        bbox_out_kernel<<<(BQ * 4 + 255) / 256, 256, 0, stream>>>(
            coord_out + (size_t)i * BQ * 4, BB3, REFINV);
    }
}
